// GINEModel_8323646619969
// MI455X (gfx1250) — compile-verified, matrix operand order checked by reading, not run on hardware
//
#include <hip/hip_runtime.h>
#include <math.h>

// ---------------------------------------------------------------------------
// Problem constants (from reference)
// ---------------------------------------------------------------------------
#define NN   20000
#define EE   320000
#define GG   64
#define DIN  128
#define HH   256
#define EF   16
#define BN_INV 0.9999950000374996f   // rsqrt(1 + 1e-5)

typedef __attribute__((ext_vector_type(16))) __bf16 v16bf;
typedef __attribute__((ext_vector_type(8)))  float  v8f;

union BfVec {
    __bf16       h[16];
    unsigned int u[8];
    v16bf        v;
};

// ---------------------------------------------------------------------------
// fp32 -> bf16 weight conversion (native cvt via __bf16 cast)
// ---------------------------------------------------------------------------
__global__ void convert_bf16_kernel(const float* __restrict__ W,
                                    __bf16* __restrict__ out, int n) {
    int i = blockIdx.x * blockDim.x + threadIdx.x;
    if (i < n) out[i] = (__bf16)W[i];
}

// ---------------------------------------------------------------------------
// Fused edge transform + message + scatter-add:
//   m = relu(x[src] + edge_attr @ We.T + be);  agg[dst] += m
// One thread per (edge, channel). 16-MAC dot; x/agg are L2-resident (20 MB
// each vs 192 MB L2), so the f32 atomics resolve in L2.
// ---------------------------------------------------------------------------
__global__ void __launch_bounds__(256)
edge_message_kernel(const float* __restrict__ x, const float* __restrict__ attr,
                    const int* __restrict__ src, const int* __restrict__ dst,
                    const float* __restrict__ We, const float* __restrict__ be,
                    float* __restrict__ agg, int C, long long total) {
    long long idx = (long long)blockIdx.x * 256 + threadIdx.x;
    if (idx >= total) return;
    int eid = (int)(idx / C);
    int c   = (int)(idx - (long long)eid * C);
    const float* a = attr + (long long)eid * EF;
    const float* w = We   + (long long)c   * EF;
    float acc = be[c];
#pragma unroll
    for (int f = 0; f < EF; ++f) acc += a[f] * w[f];
    int s = src[eid], d = dst[eid];
    float m = x[(long long)s * C + c] + acc;
    m = fmaxf(m, 0.0f);
    atomicAdd(&agg[(long long)d * C + c], m);
}

// ---------------------------------------------------------------------------
// WMMA bf16 GEMM with fused epilogue, K and A2-presence are compile-time so
// the K loop fully unrolls (K/32 back-to-back v_wmma) and no divergent
// branches surround the operand loads (WMMA requires EXEC all-ones anyway).
//   Out[m][o] = epilogue( sum_k (A1[m][k] (+A2[m][k])) * W[o][k] + bias[o] )
//   mode 0: BN+relu  -> relu((acc+ba)*BN_INV*g + bt)
//   mode 1: relu     -> relu(acc+bb)
// W supplied as bf16 row-major [256][K] so each B fragment pair is one
// aligned dword load. 8 waves/block, wave = 16x16 tile, block = 64x32.
// ---------------------------------------------------------------------------
template <int K, bool HAS_A2>
__global__ void __launch_bounds__(256)
gemm_bf16_wmma_kernel(const float* __restrict__ A1, const float* __restrict__ A2,
                      const __bf16* __restrict__ Wbf,
                      const float* __restrict__ bias,
                      const float* __restrict__ gamma, const float* __restrict__ beta,
                      float* __restrict__ Out, int M, int mode) {
    const int tid  = threadIdx.x;
    const int wave = tid >> 5;
    const int lane = tid & 31;
    const int lo   = lane & 15;
    const int hi   = lane >> 4;
    const int wm   = wave >> 1;
    const int wn   = wave & 1;
    const int m0   = blockIdx.x * 64 + wm * 16;
    const int n0   = blockIdx.y * 32 + wn * 16;

    const int mA   = m0 + lo;
    const int mC   = (mA < M) ? mA : 0;         // clamp A loads at M edge
    const int ncol = n0 + lo;                   // B column / output column

    const float*  arow1 = A1 + (long long)mC * K;
    const float*  arow2 = HAS_A2 ? (A2 + (long long)mC * K) : nullptr;
    const __bf16* brow  = Wbf + (long long)ncol * K;

    v8f acc = {0.f, 0.f, 0.f, 0.f, 0.f, 0.f, 0.f, 0.f};

#pragma unroll
    for (int kb = 0; kb < K; kb += 32) {
        BfVec a, b;
#pragma unroll
        for (int j = 0; j < 8; ++j) {
            // per-lane K index for fragment element pair (2j, 2j+1):
            //  j<4 -> k = 2j + 8*hi ;  j>=4 -> k = 16 + 2(j-4) + 8*hi
            const int kk = kb + 2 * (j & 3) + 8 * hi + ((j >= 4) ? 16 : 0);
            float2 f = *reinterpret_cast<const float2*>(arow1 + kk);
            if (HAS_A2) {
                float2 g2 = *reinterpret_cast<const float2*>(arow2 + kk);
                f.x += g2.x; f.y += g2.y;
            }
            a.h[2 * j]     = (__bf16)f.x;       // native pk cvt f32->bf16
            a.h[2 * j + 1] = (__bf16)f.y;
            b.u[j] = *reinterpret_cast<const unsigned int*>(brow + kk);
        }
        acc = __builtin_amdgcn_wmma_f32_16x16x32_bf16(
                  false, a.v, false, b.v, (short)0, acc, false, false);
    }

    const float bv = bias[ncol];
    float gm = 1.0f, bt = 0.0f;
    if (mode == 0) { gm = gamma[ncol]; bt = beta[ncol]; }
#pragma unroll
    for (int r = 0; r < 8; ++r) {
        const int row = m0 + hi * 8 + r;        // C layout: M = 8*hi + r
        float v = acc[r] + bv;
        if (mode == 0) v = v * BN_INV * gm + bt;
        v = fmaxf(v, 0.0f);
        if (row < M) Out[(long long)row * 256 + ncol] = v;
    }
}

// ---------------------------------------------------------------------------
// Pooling: atomic accumulation of h into pool[G][768] at column offset `off`
// ---------------------------------------------------------------------------
__global__ void __launch_bounds__(256)
pool_accum_kernel(const float* __restrict__ h, const int* __restrict__ batch,
                  float* __restrict__ pool, int off) {
    int idx = blockIdx.x * 256 + threadIdx.x;
    if (idx >= NN * 256) return;
    int n = idx >> 8, c = idx & 255;
    int g = batch[n];
    atomicAdd(&pool[g * 768 + off + c], h[idx]);
}

__global__ void count_kernel(const int* __restrict__ batch, float* __restrict__ cnt) {
    int i = blockIdx.x * blockDim.x + threadIdx.x;
    if (i < NN) atomicAdd(&cnt[batch[i]], 1.0f);
}

// ---------------------------------------------------------------------------
// Final MLP: one block per graph.  p = pool/cnt; h = relu(p@L1w.T + L1b);
// out = sigmoid(h@L2w.T + L2b).  64 x 768 x 768 — negligible work.
// ---------------------------------------------------------------------------
__global__ void __launch_bounds__(256)
final_mlp_kernel(const float* __restrict__ pool, const float* __restrict__ cnt,
                 const float* __restrict__ L1w, const float* __restrict__ L1b,
                 const float* __restrict__ L2w, const float* __restrict__ L2b,
                 float* __restrict__ out) {
    __shared__ float p[768];
    __shared__ float hb[768];
    __shared__ float red[256];
    const int g = blockIdx.x, tid = threadIdx.x;
    const float ic = 1.0f / fmaxf(cnt[g], 1.0f);
    for (int j = tid; j < 768; j += 256) p[j] = pool[g * 768 + j] * ic;
    __syncthreads();
    for (int o = tid; o < 768; o += 256) {
        float acc = L1b[o];
        const float* wr = L1w + (long long)o * 768;
        for (int j = 0; j < 768; ++j) acc += p[j] * wr[j];
        hb[o] = fmaxf(acc, 0.0f);
    }
    __syncthreads();
    float s = 0.0f;
    for (int j = tid; j < 768; j += 256) s += hb[j] * L2w[j];
    red[tid] = s;
    __syncthreads();
    for (int st = 128; st > 0; st >>= 1) {
        if (tid < st) red[tid] += red[tid + st];
        __syncthreads();
    }
    if (tid == 0) {
        float z = red[0] + L2b[0];
        out[g] = 1.0f / (1.0f + expf(-z));
    }
}

// ---------------------------------------------------------------------------
// Host-side orchestration
// ---------------------------------------------------------------------------
extern "C" void kernel_launch(void* const* d_in, const int* in_sizes, int n_in,
                              void* d_out, int out_size, void* d_ws, size_t ws_size,
                              hipStream_t stream) {
    const float* x    = (const float*)d_in[0];
    const float* attr = (const float*)d_in[1];
    const int*   src  = (const int*)d_in[2];
    const int*   dst  = (const int*)d_in[3];
    const int*   batch= (const int*)d_in[4];
    const float* L1w  = (const float*)d_in[29];
    const float* L1b  = (const float*)d_in[30];
    const float* L2w  = (const float*)d_in[31];
    const float* L2b  = (const float*)d_in[32];
    float* out = (float*)d_out;

    // workspace layout
    const size_t NB = (size_t)NN * 256;
    float* bufA = (float*)d_ws;            // h1 / h3
    float* bufB = bufA + NB;               // h2
    float* hmid = bufB + NB;               // intra-layer intermediate
    float* agg  = hmid + NB;               // message aggregation
    __bf16* WaBf = (__bf16*)(agg + NB);
    __bf16* WbBf = WaBf + 256 * 256;
    float* pool = (float*)(WbBf + 256 * 256);   // [64][768]
    float* cnt  = pool + GG * 768;               // [64]

    hipMemsetAsync(pool, 0, (size_t)(GG * 768 + GG) * sizeof(float), stream);
    count_kernel<<<(NN + 255) / 256, 256, 0, stream>>>(batch, cnt);

    const dim3 gemm_grid((NN + 63) / 64, 256 / 32);

    for (int layer = 0; layer < 3; ++layer) {
        const int base = 5 + 8 * layer;
        const float* We = (const float*)d_in[base + 0];
        const float* be = (const float*)d_in[base + 1];
        const float* Wa = (const float*)d_in[base + 2];
        const float* ba = (const float*)d_in[base + 3];
        const float* gm = (const float*)d_in[base + 4];
        const float* bt = (const float*)d_in[base + 5];
        const float* Wb = (const float*)d_in[base + 6];
        const float* bb = (const float*)d_in[base + 7];

        const int C = (layer == 0) ? DIN : HH;
        const float* xin = (layer == 0) ? x : ((layer == 1) ? bufA : bufB);
        float* hout      = (layer == 1) ? bufB : bufA;   // h1->A, h2->B, h3->A

        // 1) zero aggregation buffer, run fused edge message/scatter
        hipMemsetAsync(agg, 0, (size_t)NN * C * sizeof(float), stream);
        long long totalEC = (long long)EE * C;
        edge_message_kernel<<<(unsigned)((totalEC + 255) / 256), 256, 0, stream>>>(
            xin, attr, src, dst, We, be, agg, C, totalEC);

        // 2) convert weights to bf16
        convert_bf16_kernel<<<(256 * C + 255) / 256, 256, 0, stream>>>(Wa, WaBf, 256 * C);
        convert_bf16_kernel<<<(256 * 256 + 255) / 256, 256, 0, stream>>>(Wb, WbBf, 256 * 256);

        // 3) h_mid = relu(BN((x+agg)@Wa.T + ba))   [WMMA, fused x+agg]
        if (C == DIN) {
            gemm_bf16_wmma_kernel<DIN, true><<<gemm_grid, 256, 0, stream>>>(
                xin, agg, WaBf, ba, gm, bt, hmid, NN, /*mode=*/0);
        } else {
            gemm_bf16_wmma_kernel<HH, true><<<gemm_grid, 256, 0, stream>>>(
                xin, agg, WaBf, ba, gm, bt, hmid, NN, /*mode=*/0);
        }

        // 4) h_out = relu(h_mid@Wb.T + bb)          [WMMA]
        gemm_bf16_wmma_kernel<HH, false><<<gemm_grid, 256, 0, stream>>>(
            hmid, nullptr, WbBf, bb, nullptr, nullptr, hout, NN, /*mode=*/1);

        // 5) accumulate into pooled features at column offset 256*layer
        pool_accum_kernel<<<(NN * 256 + 255) / 256, 256, 0, stream>>>(
            hout, batch, pool, 256 * layer);
    }

    final_mlp_kernel<<<GG, 256, 0, stream>>>(pool, cnt, L1w, L1b, L2w, L2b, out);
    (void)in_sizes; (void)n_in; (void)out_size; (void)ws_size;
}
